// ParallelTransformerAEP_64811056497152
// MI455X (gfx1250) — compile-verified, matrix operand order checked by reading, not run on hardware
//
#include <hip/hip_runtime.h>
#include <hip/hip_bf16.h>

// ---------------- model constants ----------------
#define NUM_LAYERS 4
#define HEADS      8
#define DH         64
#define DIM        512
#define FF_INNER   2048
#define Bsz        32
#define SEQ        512
#define ATTN_INNER (HEADS * DH)                 // 512
#define FUSED      (ATTN_INNER + DH + DH + 2 * FF_INNER)   // 4736
#define MROWS      (Bsz * SEQ)                  // 16384

typedef __attribute__((ext_vector_type(16))) __bf16 v16bf;
typedef __attribute__((ext_vector_type(8)))  float  v8f;
typedef int v4i __attribute__((vector_size(16)));   // matches builtin prototype

// ---- CDNA5 async global->LDS path (guarded; sync fallback) ----
#if defined(__has_builtin)
#  if __has_builtin(__builtin_amdgcn_global_load_async_to_lds_b128) && \
      __has_builtin(__builtin_amdgcn_s_wait_asynccnt)
#    define USE_ASYNC_LDS 1
#  endif
#endif
#ifndef USE_ASYNC_LDS
#  define USE_ASYNC_LDS 0
#endif

__device__ __forceinline__ void copy16_to_lds(const __bf16* g, __bf16* l) {
#if USE_ASYNC_LDS
    __builtin_amdgcn_global_load_async_to_lds_b128(
        (__attribute__((address_space(1))) v4i*)g,
        (__attribute__((address_space(3))) v4i*)l, 0, 0);
#else
    *(uint4*)l = *(const uint4*)g;
#endif
}
__device__ __forceinline__ void async_fence() {
#if USE_ASYNC_LDS
    __builtin_amdgcn_s_wait_asynccnt(0);
#endif
}

__device__ __forceinline__ v8f zero8() {
    v8f z;
#pragma unroll
    for (int i = 0; i < 8; ++i) z[i] = 0.f;
    return z;
}

__device__ __forceinline__ v8f wmma_bf16(v16bf a, v16bf b, v8f c) {
    // D = A(16x32 bf16) * B(32x16 bf16) + C(16x16 f32)
    return __builtin_amdgcn_wmma_f32_16x16x32_bf16(false, a, false, b, (short)0, c,
                                                   false, false);
}

// 16x32 bf16 A-layout fragment (also used for B tiles stored K-major):
// lanes 0-15: row = lane,    K in {0..7, 16..23}
// lanes16-31: row = lane-16, K in {8..15, 24..31}
// K pairs are contiguous 32-bit words -> compiler merges into 2x ds_load_b128.
__device__ __forceinline__ v16bf load_frag(const __bf16* base, int stride, int row,
                                           int kofs, int lane) {
    union { v16bf v; unsigned int u[8]; } f;
    const int kb = (lane < 16) ? 0 : 8;
    const __bf16* rp = base + (size_t)row * stride + kofs;
#pragma unroll
    for (int p = 0; p < 8; ++p) {
        int k = (p < 4) ? (kb + 2 * p) : (kb + 16 + 2 * (p - 4));
        f.u[p] = *(const unsigned int*)(rp + k);
    }
    return f.v;
}

// ---------------- weight convert f32 -> bf16 ----------------
__global__ void cvt_bf16_kernel(const float* __restrict__ src, __bf16* __restrict__ dst,
                                int n) {
    int i = blockIdx.x * blockDim.x + threadIdx.x;
    for (; i < n; i += gridDim.x * blockDim.x) dst[i] = (__bf16)src[i];
}

// ---------------- embeddings + wide features -> x ----------------
__global__ void embed_kernel(const int* __restrict__ page_in, const int* __restrict__ item_in,
                             const int* __restrict__ item_meta_in,
                             const int* __restrict__ page_meta_in,
                             const float* __restrict__ pwide, const float* __restrict__ iwide,
                             const float* __restrict__ pet,  const float* __restrict__ pmet,
                             const float* __restrict__ iet,  const float* __restrict__ imet,
                             const float* __restrict__ ipet,
                             const float* __restrict__ pW, const float* __restrict__ pb,
                             const float* __restrict__ pg, const float* __restrict__ pbb,
                             const float* __restrict__ iW, const float* __restrict__ ib,
                             const float* __restrict__ ig, const float* __restrict__ ibb,
                             float* __restrict__ x) {
    const int token = blockIdx.x;
    const int b = token / SEQ, s = token % SEQ;
    const int t = threadIdx.x;
    __shared__ float wnp[8], wni[8];
    const float invs = 0.9999950000374997f;  // 1/sqrt(1+1e-5)
    if (t < 8)       wnp[t]     = pwide[((size_t)b * 8 + t) * SEQ + s] * invs * pg[t] + pbb[t];
    else if (t < 16) wni[t - 8] = iwide[((size_t)b * 8 + (t - 8)) * SEQ + s] * invs * ig[t - 8] + ibb[t - 8];
    __syncthreads();
    const int pidx = page_in[token], iidx = item_in[token];
    const int pmi = page_meta_in[token], imi = item_meta_in[token];
    for (int c = t; c < DIM; c += blockDim.x) {
        float pv, iv;
        if (c < 384) pv = pet[(size_t)pidx * 384 + c];
        else if (c < 448) pv = pmet[(size_t)pmi * 64 + (c - 384)];
        else {
            int j = c - 448;
            float a = pb[j];
#pragma unroll
            for (int f = 0; f < 8; ++f) a += wnp[f] * pW[f * 64 + j];
            pv = (a > 0.f) ? a : 0.2f * a;
        }
        if (c < 320) iv = iet[(size_t)iidx * 320 + c];
        else if (c < 384) iv = imet[(size_t)imi * 64 + (c - 320)];
        else if (c < 448) iv = ipet[(size_t)iidx * 64 + (c - 384)];
        else {
            int j = c - 448;
            float a = ib[j];
#pragma unroll
            for (int f = 0; f < 8; ++f) a += wni[f] * iW[f * 64 + j];
            iv = (a > 0.f) ? a : 0.2f * a;
        }
        x[(size_t)token * DIM + c] = pv * iv;
    }
}

// ---------------- layernorm (per token) -> bf16 ----------------
__global__ void ln_kernel(const float* __restrict__ x, const float* __restrict__ g,
                          __bf16* __restrict__ xn) {
    const int token = blockIdx.x;
    const int t = threadIdx.x;
    __shared__ float red[256];
    const float* xr = x + (size_t)token * DIM;
    float a0 = xr[t], a1 = xr[t + 256];
    red[t] = a0 + a1;
    __syncthreads();
    for (int off = 128; off > 0; off >>= 1) {
        if (t < off) red[t] += red[t + off];
        __syncthreads();
    }
    const float mu = red[0] * (1.f / DIM);
    __syncthreads();
    float d0 = a0 - mu, d1 = a1 - mu;
    red[t] = d0 * d0 + d1 * d1;
    __syncthreads();
    for (int off = 128; off > 0; off >>= 1) {
        if (t < off) red[t] += red[t + off];
        __syncthreads();
    }
    const float rstd = rsqrtf(red[0] * (1.f / DIM) + 1e-5f);
    xn[(size_t)token * DIM + t]       = (__bf16)(d0 * rstd * g[t]);
    xn[(size_t)token * DIM + t + 256] = (__bf16)(d1 * rstd * g[t + 256]);
}

// ---------------- generic bf16 WMMA GEMM, double-buffered async staging ----------------
// C[M,N] = A[M,K](bf16,row-major) * Bw[K,N](bf16,row-major)
// MODE 0: store f32   MODE 1: store bf16   MODE 2: f32 +=
#define BM 128
#define BN 128
#define BK 32
template <int MODE>
__global__ __launch_bounds__(256) void gemm_bf16_kernel(
        const __bf16* __restrict__ A, int lda, const __bf16* __restrict__ Bw, int ldb,
        void* __restrict__ Cv, int ldc, int K) {
    __shared__ __bf16 As[2][BM * BK];     // [m][k]  ping-pong
    __shared__ __bf16 Bs[2][BN * BK];     // transposed [n][k]  ping-pong
    const int t = threadIdx.x;
    const int wave = t >> 5, lane = t & 31;
    const int wm = wave >> 1, wn = wave & 1;          // 4x2 wave grid, 32x64 per wave
    const int bm0 = blockIdx.y * BM, bn0 = blockIdx.x * BN;

    v8f acc[2][4];
#pragma unroll
    for (int i = 0; i < 2; ++i)
#pragma unroll
        for (int j = 0; j < 4; ++j) acc[i][j] = zero8();

    const int a_kc = (t & 3) * 8, a_row = t >> 2;
    const int b_k = t >> 3, b_n0 = (t & 7) * 16;

    // stage K-tile k0 into buffer buf: A via async DMA, B via transposing stores
    auto stage_tile = [&](int k0, int buf) {
#pragma unroll
        for (int pass = 0; pass < 2; ++pass) {
            int mm = pass * 64 + a_row;
            copy16_to_lds(A + (size_t)(bm0 + mm) * lda + k0 + a_kc,
                          &As[buf][mm * BK + a_kc]);
        }
        union { uint4 u; __bf16 h[8]; } bb0, bb1;
        const __bf16* bp = Bw + (size_t)(k0 + b_k) * ldb + bn0 + b_n0;
        bb0.u = *(const uint4*)(bp);
        bb1.u = *(const uint4*)(bp + 8);
#pragma unroll
        for (int j = 0; j < 8; ++j) {
            Bs[buf][(b_n0 + j) * BK + b_k]     = bb0.h[j];
            Bs[buf][(b_n0 + 8 + j) * BK + b_k] = bb1.h[j];
        }
    };

    const int nT = K / BK;
    stage_tile(0, 0);
    for (int ti = 0; ti < nT; ++ti) {
        async_fence();            // my DMA for buffer `cur` landed
        __syncthreads();          // everyone's staging of `cur` visible; `cur^1` free
        const int cur = ti & 1;
        if (ti + 1 < nT) {
            stage_tile((ti + 1) * BK, cur ^ 1);   // overlap next DMA with WMMAs
            if (ti + 2 < nT)
                __builtin_prefetch(Bw + (size_t)((ti + 2) * BK + b_k) * ldb + bn0 + b_n0, 0, 1);
        }
        const __bf16* Asb = &As[cur][0];
        const __bf16* Bsb = &Bs[cur][0];
        v16bf af[2], bfr[4];
#pragma unroll
        for (int mi = 0; mi < 2; ++mi)
            af[mi] = load_frag(Asb, BK, wm * 32 + mi * 16 + (lane & 15), 0, lane);
#pragma unroll
        for (int ni = 0; ni < 4; ++ni)
            bfr[ni] = load_frag(Bsb, BK, wn * 64 + ni * 16 + (lane & 15), 0, lane);
#pragma unroll
        for (int mi = 0; mi < 2; ++mi)
#pragma unroll
            for (int ni = 0; ni < 4; ++ni)
                acc[mi][ni] = wmma_bf16(af[mi], bfr[ni], acc[mi][ni]);
    }

    // straight-line epilogue (C layout: VGPR r -> row +r (+8 for lanes>=16), col=lane&15)
#pragma unroll
    for (int mi = 0; mi < 2; ++mi)
#pragma unroll
        for (int ni = 0; ni < 4; ++ni)
#pragma unroll
            for (int r = 0; r < 8; ++r) {
                int row = bm0 + wm * 32 + mi * 16 + r + ((lane < 16) ? 0 : 8);
                int col = bn0 + wn * 64 + ni * 16 + (lane & 15);
                size_t idx = (size_t)row * ldc + col;
                float v = acc[mi][ni][r];
                if constexpr (MODE == 0)      ((float*)Cv)[idx] = v;
                else if constexpr (MODE == 1) ((__bf16*)Cv)[idx] = (__bf16)v;
                else                          ((float*)Cv)[idx] += v;
            }
}

// ---------------- RoPE + split q/k/v from fused output ----------------
__global__ void rope_kernel(const __bf16* __restrict__ p, __bf16* __restrict__ qb,
                            __bf16* __restrict__ kb, __bf16* __restrict__ vb) {
    const int token = blockIdx.x;
    const int b = token / SEQ, s = token % SEQ;
    const int t = threadIdx.x;
    const __bf16* pr = p + (size_t)token * FUSED;
    for (int c = t; c < ATTN_INNER; c += blockDim.x) {
        int h = c >> 6, d = c & 63, dr = d & 31;
        float invf = __powf(10000.f, -(float)dr * (1.f / 32.f));
        float sn, cs;
        __sincosf((float)s * invf, &sn, &cs);
        float v0 = (float)pr[c];
        float v1 = (float)pr[(h << 6) | ((d < 32) ? d + 32 : d - 32)];
        float r = (d < 32) ? (v0 * cs - v1 * sn) : (v0 * cs + v1 * sn);
        qb[(((size_t)(b * HEADS + h)) * SEQ + s) * DH + d] = (__bf16)(r * 0.125f);
    }
    if (t < DH) {
        int d = t, dr = d & 31;
        float invf = __powf(10000.f, -(float)dr * (1.f / 32.f));
        float sn, cs;
        __sincosf((float)s * invf, &sn, &cs);
        float v0 = (float)pr[ATTN_INNER + d];
        float v1 = (float)pr[ATTN_INNER + ((d < 32) ? d + 32 : d - 32)];
        float r = (d < 32) ? (v0 * cs - v1 * sn) : (v0 * cs + v1 * sn);
        kb[(size_t)token * DH + d] = (__bf16)r;
    } else if (t < 2 * DH) {
        int d = t - DH;
        vb[(size_t)token * DH + d] = pr[ATTN_INNER + DH + d];
    }
}

// ---------------- SwiGLU: h = silu(gate) * xx ----------------
__global__ void silu_kernel(const __bf16* __restrict__ p, __bf16* __restrict__ hbuf) {
    const int total = MROWS * FF_INNER;
    int i = blockIdx.x * blockDim.x + threadIdx.x;
    for (; i < total; i += gridDim.x * blockDim.x) {
        int row = i >> 11, c = i & (FF_INNER - 1);
        float xx = (float)p[(size_t)row * FUSED + ATTN_INNER + 2 * DH + c];
        float g  = (float)p[(size_t)row * FUSED + ATTN_INNER + 2 * DH + FF_INNER + c];
        float sg = g / (1.f + __expf(-g));
        hbuf[i] = (__bf16)(sg * xx);
    }
}

// ---------------- flash attention (MQA), WMMA + double-buffered K/V ----------------
// grid = B*H*4, each workgroup: 8 waves x 16 query rows = 128 rows
__global__ __launch_bounds__(256) void attn_kernel(
        const __bf16* __restrict__ qb, const __bf16* __restrict__ kb,
        const __bf16* __restrict__ vb, const int* __restrict__ vl_in,
        __bf16* __restrict__ ao) {
    __shared__ __bf16 Kt[2][32 * 64];       // [key][dh]       ping-pong
    __shared__ __bf16 Vt[2][64 * 32];       // transposed [dh][key] ping-pong
    __shared__ __bf16 Ps[8][16 * 32];       // per-wave P tile [row][key]

    const int blk = blockIdx.x;
    const int b = blk >> 5;                 // H*4 = 32
    const int h = (blk >> 2) & 7;
    const int quarter = blk & 3;
    const int t = threadIdx.x, wave = t >> 5, lane = t & 31;
    const int qrow0 = quarter * 128 + wave * 16;
    const int vl = vl_in[b];

    const __bf16* qbase = qb + (((size_t)(b * HEADS + h)) * SEQ + qrow0) * DH;
    v16bf qf[2];
    qf[0] = load_frag(qbase, DH, lane & 15, 0, lane);
    qf[1] = load_frag(qbase, DH, lane & 15, 32, lane);

    v8f o[4];
#pragma unroll
    for (int j = 0; j < 4; ++j) o[j] = zero8();
    float m[8], l[8];
#pragma unroll
    for (int r = 0; r < 8; ++r) { m[r] = -1e30f; l[r] = 0.f; }

    const int ld_key = t >> 3, ld_d0 = (t & 7) * 8;
    // stage a 32-key block: K via async DMA, V^T via transposing stores
    auto stage_kv = [&](int kb0, int buf) {
        copy16_to_lds(kb + ((size_t)(b * SEQ + kb0 + ld_key)) * DH + ld_d0,
                      &Kt[buf][ld_key * 64 + ld_d0]);
        union { uint4 u; __bf16 hh[8]; } vv;
        vv.u = *(const uint4*)(vb + ((size_t)(b * SEQ + kb0 + ld_key)) * DH + ld_d0);
#pragma unroll
        for (int j = 0; j < 8; ++j) Vt[buf][(ld_d0 + j) * 32 + ld_key] = vv.hh[j];
    };

    const int nBlk = SEQ / 32;
    stage_kv(0, 0);
    for (int bi = 0; bi < nBlk; ++bi) {
        async_fence();
        __syncthreads();
        const int cur = bi & 1;
        if (bi + 1 < nBlk) stage_kv((bi + 1) * 32, cur ^ 1);  // overlap next K/V DMA
        const int kb0 = bi * 32;
        const __bf16* Ktb = &Kt[cur][0];
        const __bf16* Vtb = &Vt[cur][0];

        // S = Q (16x64) @ K^T (64x32) -> two 16x16 tiles
        v8f sacc[2];
        sacc[0] = zero8(); sacc[1] = zero8();
#pragma unroll
        for (int ks = 0; ks < 2; ++ks) {
            v16bf kf0 = load_frag(Ktb, 64, (lane & 15), ks * 32, lane);
            v16bf kf1 = load_frag(Ktb, 64, 16 + (lane & 15), ks * 32, lane);
            sacc[0] = wmma_bf16(qf[ks], kf0, sacc[0]);
            sacc[1] = wmma_bf16(qf[ks], kf1, sacc[1]);
        }
        // key-pad mask: column index = lane&15 (uniform across accum VGPRs)
#pragma unroll
        for (int ni = 0; ni < 2; ++ni) {
            int key = kb0 + ni * 16 + (lane & 15);
            if (key >= vl) {
#pragma unroll
                for (int r = 0; r < 8; ++r) sacc[ni][r] = -1e30f;
            }
        }
        // online softmax update (row reductions across 16-lane groups)
        float pe[2][8], scale[8];
#pragma unroll
        for (int r = 0; r < 8; ++r) {
            float rv = fmaxf(sacc[0][r], sacc[1][r]);
            rv = fmaxf(rv, __shfl_xor(rv, 1));
            rv = fmaxf(rv, __shfl_xor(rv, 2));
            rv = fmaxf(rv, __shfl_xor(rv, 4));
            rv = fmaxf(rv, __shfl_xor(rv, 8));
            float mn = fmaxf(m[r], rv);
            float sc = __expf(m[r] - mn);
            float p0 = __expf(sacc[0][r] - mn);
            float p1 = __expf(sacc[1][r] - mn);
            float rs = p0 + p1;
            rs += __shfl_xor(rs, 1);
            rs += __shfl_xor(rs, 2);
            rs += __shfl_xor(rs, 4);
            rs += __shfl_xor(rs, 8);
            l[r] = l[r] * sc + rs;
            m[r] = mn;
            scale[r] = sc;
            pe[0][r] = p0;
            pe[1][r] = p1;
        }
#pragma unroll
        for (int j = 0; j < 4; ++j)
#pragma unroll
            for (int r = 0; r < 8; ++r) o[j][r] *= scale[r];

        // re-layout P through per-wave LDS: C-layout -> A-layout (16x32)
        __bf16* pw = &Ps[wave][0];
#pragma unroll
        for (int ni = 0; ni < 2; ++ni)
#pragma unroll
            for (int r = 0; r < 8; ++r)
                pw[(r + ((lane < 16) ? 0 : 8)) * 32 + ni * 16 + (lane & 15)] =
                    (__bf16)pe[ni][r];
        v16bf pf = load_frag(pw, 32, lane & 15, 0, lane);
#pragma unroll
        for (int j = 0; j < 4; ++j) {
            v16bf vf = load_frag(Vtb, 32, j * 16 + (lane & 15), 0, lane);
            o[j] = wmma_bf16(pf, vf, o[j]);
        }
    }

    float invl[8];
#pragma unroll
    for (int r = 0; r < 8; ++r) invl[r] = 1.f / l[r];
#pragma unroll
    for (int j = 0; j < 4; ++j)
#pragma unroll
        for (int r = 0; r < 8; ++r) {
            int row = qrow0 + r + ((lane < 16) ? 0 : 8);
            int col = h * DH + j * 16 + (lane & 15);
            ao[((size_t)(b * SEQ + row)) * ATTN_INNER + col] = (__bf16)(o[j][r] * invl[r]);
        }
}

// ---------------- residual add ----------------
__global__ void add_kernel(float* __restrict__ x, const float* __restrict__ y, int n) {
    int i = blockIdx.x * blockDim.x + threadIdx.x;
    for (; i < n; i += gridDim.x * blockDim.x) x[i] += y[i];
}

// ---------------- final mean/max pooling ----------------
__global__ void pool_kernel(const float* __restrict__ x, float* __restrict__ out) {
    const int b = blockIdx.x;
    const int d = threadIdx.x;  // 512
    float s = 0.f, mx = -3.4e38f;
    for (int i = 0; i < SEQ; ++i) {
        float v = x[((size_t)(b * SEQ + i)) * DIM + d];
        s += v;
        mx = fmaxf(mx, v);
    }
    out[(size_t)b * 1024 + d] = s * (1.f / SEQ);
    out[(size_t)b * 1024 + DIM + d] = mx;
}

// ---------------- launch ----------------
extern "C" void kernel_launch(void* const* d_in, const int* in_sizes, int n_in,
                              void* d_out, int out_size, void* d_ws, size_t ws_size,
                              hipStream_t stream) {
    (void)in_sizes; (void)n_in; (void)out_size; (void)ws_size;
    const int*   page_in      = (const int*)d_in[0];
    const int*   item_in      = (const int*)d_in[1];
    const int*   item_meta_in = (const int*)d_in[2];
    const int*   vl_in        = (const int*)d_in[3];
    const int*   page_meta_in = (const int*)d_in[4];
    const float* pwide        = (const float*)d_in[5];
    const float* iwide        = (const float*)d_in[6];
    const float* pet          = (const float*)d_in[7];
    const float* pmet         = (const float*)d_in[8];
    const float* iet          = (const float*)d_in[9];
    const float* imet         = (const float*)d_in[10];
    const float* ipet         = (const float*)d_in[11];
    const float* pW           = (const float*)d_in[12];
    const float* pb           = (const float*)d_in[13];
    const float* pg           = (const float*)d_in[14];
    const float* pbb          = (const float*)d_in[15];
    const float* iW           = (const float*)d_in[16];
    const float* ib           = (const float*)d_in[17];
    const float* ig           = (const float*)d_in[18];
    const float* ibb          = (const float*)d_in[19];
    const float* norm_g       = (const float*)d_in[20];
    const float* fused_W      = (const float*)d_in[21];
    const float* attn_out_W   = (const float*)d_in[22];
    const float* ff_out_W     = (const float*)d_in[23];
    float* out = (float*)d_out;

    // ---- workspace layout (all sizes multiples of 256B) ----
    char* ws = (char*)d_ws;
    size_t off = 0;
    __bf16* wF = (__bf16*)(ws + off); off += (size_t)NUM_LAYERS * DIM * FUSED * 2;
    __bf16* wA = (__bf16*)(ws + off); off += (size_t)NUM_LAYERS * ATTN_INNER * DIM * 2;
    __bf16* wO = (__bf16*)(ws + off); off += (size_t)NUM_LAYERS * FF_INNER * DIM * 2;
    float*  x  = (float*) (ws + off); off += (size_t)MROWS * DIM * 4;
    __bf16* xn = (__bf16*)(ws + off); off += (size_t)MROWS * DIM * 2;
    __bf16* p  = (__bf16*)(ws + off); off += (size_t)MROWS * FUSED * 2;
    __bf16* qb = (__bf16*)(ws + off); off += (size_t)MROWS * ATTN_INNER * 2;
    __bf16* kb = (__bf16*)(ws + off); off += (size_t)MROWS * DH * 2;
    __bf16* vb = (__bf16*)(ws + off); off += (size_t)MROWS * DH * 2;
    __bf16* hb = (__bf16*)(ws + off); off += (size_t)MROWS * FF_INNER * 2;
    __bf16* ao = (__bf16*)(ws + off); off += (size_t)MROWS * ATTN_INNER * 2;
    float*  y  = (float*) (ws + off); off += (size_t)MROWS * DIM * 4;

    // weights -> bf16 (every call; deterministic)
    cvt_bf16_kernel<<<4096, 256, 0, stream>>>(fused_W, wF, NUM_LAYERS * DIM * FUSED);
    cvt_bf16_kernel<<<2048, 256, 0, stream>>>(attn_out_W, wA, NUM_LAYERS * ATTN_INNER * DIM);
    cvt_bf16_kernel<<<2048, 256, 0, stream>>>(ff_out_W, wO, NUM_LAYERS * FF_INNER * DIM);

    embed_kernel<<<MROWS, 256, 0, stream>>>(page_in, item_in, item_meta_in, page_meta_in,
                                            pwide, iwide, pet, pmet, iet, imet, ipet,
                                            pW, pb, pg, pbb, iW, ib, ig, ibb, x);

    for (int l = 0; l < NUM_LAYERS; ++l) {
        ln_kernel<<<MROWS, 256, 0, stream>>>(x, norm_g + (size_t)l * DIM, xn);

        // p = xn @ fused_W[l]  (16384 x 4736), bf16 out
        gemm_bf16_kernel<1><<<dim3(FUSED / BN, MROWS / BM), 256, 0, stream>>>(
            xn, DIM, wF + (size_t)l * DIM * FUSED, FUSED, p, FUSED, DIM);

        rope_kernel<<<MROWS, 256, 0, stream>>>(p, qb, kb, vb);
        silu_kernel<<<16384, 256, 0, stream>>>(p, hb);

        attn_kernel<<<Bsz * HEADS * 4, 256, 0, stream>>>(qb, kb, vb, vl_in, ao);

        // y = ao @ attn_out_W[l]  (f32 store)
        gemm_bf16_kernel<0><<<dim3(DIM / BN, MROWS / BM), 256, 0, stream>>>(
            ao, ATTN_INNER, wA + (size_t)l * ATTN_INNER * DIM, DIM, y, DIM, ATTN_INNER);
        // y += hb @ ff_out_W[l]
        gemm_bf16_kernel<2><<<dim3(DIM / BN, MROWS / BM), 256, 0, stream>>>(
            hb, FF_INNER, wO + (size_t)l * FF_INNER * DIM, DIM, y, DIM, FF_INNER);

        add_kernel<<<8192, 256, 0, stream>>>(x, y, MROWS * DIM);
    }

    pool_kernel<<<Bsz, DIM, 0, stream>>>(x, out);
}
